// NeRF_9629316678123
// MI455X (gfx1250) — matrix operand order repaired and checked by reading, not compiled, not run on hardware
//
#include <hip/hip_runtime.h>

typedef __attribute__((ext_vector_type(16))) _Float16 v16h;
typedef __attribute__((ext_vector_type(8)))  _Float16 v8h;
typedef __attribute__((ext_vector_type(8)))  float    v8f;
typedef __attribute__((ext_vector_type(4)))  float    v4f;

union U16 { v16h v; v8h h8[2]; _Float16 h[16]; unsigned u[8]; };
union C8  { v8f v; v4f f4[2]; float f[8]; };
union D4  { v8h v; _Float16 h[8]; unsigned u[4]; };   // f16 C/D tile (16x16, 2 rows/VGPR)

#define HASH_T   524288u          // 2^19 table entries per level
#define HASH_TM  (HASH_T - 1u)

// floor(16 * exp(k * ln(128)/15)), double-precision (matches numpy)
constexpr int RES_TAB[16] = {16,22,30,42,58,80,111,153,212,294,406,561,776,1072,1482,2048};

__device__ __forceinline__ v8f wmma_f32(v16h a, v16h b, v8f c) {
  return __builtin_amdgcn_wmma_f32_16x16x32_f16(false, a, false, b, (short)0, c, false, false);
}
__device__ __forceinline__ v8h wmma_f16(v16h a, v16h b, v8h c) {
  return __builtin_amdgcn_wmma_f16_16x16x32_f16(false, a, false, b, (short)0, c, false, false);
}

// Pure-VALU lane <-> lane^16 exchange (wave32 half swap).
__device__ __forceinline__ unsigned xor16u(unsigned v) {
#if __has_builtin(__builtin_amdgcn_permlanex16)
  return (unsigned)__builtin_amdgcn_permlanex16((int)v, (int)v,
                                                0x76543210, 0xfedcba98, false, false);
#else
  return (unsigned)__shfl_xor((int)v, 16, 32);
#endif
}
__device__ __forceinline__ float xor16f(float v) {
  unsigned r = xor16u(__float_as_uint(v));
  return __uint_as_float(r);
}

// Single-instruction packed f16 relu (asm so InstCombine can't re-expand).
__device__ __forceinline__ unsigned relu2h(unsigned x) {
  unsigned r;
  asm("v_pk_max_num_f16 %0, %1, 0" : "=v"(r) : "v"(x));
  return r;
}

// Stage weight matrix W (fi x fo, row-major, x@W convention) into LDS in
// A-fragment order: frag = mt*nK + kt, 512 halves per frag (32 lanes x 16 h).
// A element (M,K): lane<16 -> M=lane, K in {0-15}; lane>=16 -> K in {16-31}.
__device__ __forceinline__ void stage_w(const float* __restrict__ W, int fi, int fo,
                                        _Float16* __restrict__ dst, int tid, int nthr) {
  const int nK = (fi + 31) >> 5;
  const int nM = (fo + 15) >> 4;
  const int total = nM * nK * 512;
  for (int idx = tid; idx < total; idx += nthr) {
    const int frag = idx >> 9;
    const int lane = (idx >> 4) & 31;
    const int h    = idx & 15;
    const int kt = frag % nK, mt = frag / nK;
    const int M = mt * 16 + (lane & 15);
    const int K = kt * 32 + ((lane & 16) ? 16 : 0) + h;
    float v = (M < fo && K < fi) ? W[K * fo + M] : 0.f;
    dst[idx] = (_Float16)v;
  }
}

// f32-accumulate layer (used for the two output layers feeding exp/sigmoid).
template<int NK, int NM>
__device__ __forceinline__ void run_layer_f(const _Float16* __restrict__ wl,
                                            const float* __restrict__ bl,
                                            const U16* bf, C8* ct, int lane) {
#pragma unroll
  for (int mt = 0; mt < NM; ++mt) {
    C8 c;
    const float* bp = bl + mt * 16 + ((lane & 16) ? 8 : 0);
    c.f4[0] = *(const v4f*)(bp);
    c.f4[1] = *(const v4f*)(bp + 4);
#pragma unroll
    for (int kt = 0; kt < NK; ++kt) {
      U16 a;
      const _Float16* p = wl + (mt * NK + kt) * 512 + lane * 16;
      a.h8[0] = *(const v8h*)(p);
      a.h8[1] = *(const v8h*)(p + 8);
      c.v = wmma_f32(a.v, bf[kt].v, c.v);
    }
    ct[mt] = c;
  }
}

// f16-accumulate hidden layer. f16 C/D tile: lo lanes hold M=0..7 (2 rows
// packed per VGPR), hi lanes M=8..15. Bias init: plain contiguous f16 bias,
// one ds_load_b128 per tile (lo: [mt*16+0..7], hi: [mt*16+8..15]).
template<int NK, int NM>
__device__ __forceinline__ void run_layer_h(const _Float16* __restrict__ wl,
                                            const _Float16* __restrict__ blh,
                                            const U16* bf, D4* dt, int lane) {
#pragma unroll
  for (int mt = 0; mt < NM; ++mt) {
    D4 c;
    const _Float16* bp = blh + mt * 16 + ((lane & 16) ? 8 : 0);
    c.v = *(const v8h*)(bp);              // ds_load_b128
#pragma unroll
    for (int kt = 0; kt < NK; ++kt) {
      U16 a;
      const _Float16* p = wl + (mt * NK + kt) * 512 + lane * 16;
      a.h8[0] = *(const v8h*)(p);
      a.h8[1] = *(const v8h*)(p + 8);
      c.v = wmma_f16(a.v, bf[kt].v, c.v);
    }
    dt[mt] = c;
  }
}

// Packed repack: f16 D tiles -> next layer's B fragments, with packed relu.
// B VGPR r = K(2r,2r+1); D tile VGPR r = M(2r,2r+1) [+8 on hi lanes].
// Per lane pair: lo keeps d0, needs d0's M8..15 from hi; hi keeps d1, needs
// d1's M0..7 from lo  => ONE packed exchange per VGPR.
template<int NKT>
__device__ __forceinline__ void repack_relu_h(const D4* dt, U16* bf, bool lo) {
#pragma unroll
  for (int kt = 0; kt < NKT; ++kt) {
#pragma unroll
    for (int r = 0; r < 4; ++r) {
      unsigned d0 = dt[2*kt].u[r], d1 = dt[2*kt + 1].u[r];
      unsigned own  = lo ? d0 : d1;
      unsigned send = lo ? d1 : d0;
      unsigned recv = xor16u(send);
      unsigned ro = relu2h(own);
      unsigned rr = relu2h(recv);
      bf[kt].u[r]     = lo ? ro : rr;
      bf[kt].u[4 + r] = lo ? rr : ro;
    }
  }
}

__global__ __launch_bounds__(256) void nerf_fused_kernel(
    const float* __restrict__ xyz, const float* __restrict__ tables,
    const float* __restrict__ w1,  const float* __restrict__ b1,
    const float* __restrict__ w2,  const float* __restrict__ b2,
    const float* __restrict__ w3,  const float* __restrict__ b3,
    const float* __restrict__ cw1, const float* __restrict__ cb1,
    const float* __restrict__ cw2, const float* __restrict__ cb2,
    const float* __restrict__ cw3, const float* __restrict__ cb3,
    const float* __restrict__ cw4, const float* __restrict__ cb4,
    float* __restrict__ out, int N)
{
  // LDS: weights in A-frag order + biases (f32 for f32-acc layers, f16 packed)
  __shared__ __align__(16) _Float16 sW[18432];   // 36 KB
  __shared__ __align__(16) float    sB[352];     // 1.4 KB
  __shared__ __align__(16) _Float16 sBH[352];    // 0.7 KB

  const int tid  = threadIdx.x;
  const int nthr = blockDim.x;

  stage_w(w1,  32, 64, sW + 0,     tid, nthr);   // 4 frags
  stage_w(w2,  64, 64, sW + 2048,  tid, nthr);   // 8 frags
  stage_w(w3,  64, 16, sW + 6144,  tid, nthr);   // 2 frags
  stage_w(cw1, 16, 64, sW + 7168,  tid, nthr);   // 4 frags (K padded to 32)
  stage_w(cw2, 64, 64, sW + 9216,  tid, nthr);   // 8 frags
  stage_w(cw3, 64, 64, sW + 13312, tid, nthr);   // 8 frags
  stage_w(cw4, 64, 3,  sW + 17408, tid, nthr);   // 2 frags (M padded to 16)
  for (int i = tid; i < 352; i += nthr) {
    float v = 0.f;
    if      (i <  64) v = b1[i];
    else if (i < 128) v = b2[i - 64];
    else if (i < 144) v = b3[i - 128];
    else if (i < 208) v = cb1[i - 144];
    else if (i < 272) v = cb2[i - 208];
    else if (i < 336) v = cb3[i - 272];
    else if (i < 339) v = cb4[i - 336];
    sB[i]  = v;
    sBH[i] = (_Float16)v;
  }
  __syncthreads();

  const int  lane   = tid & 31;
  const bool lo     = (lane & 16) == 0;
  const int  wave   = blockIdx.x * (nthr >> 5) + (tid >> 5);
  const int  nwaves = gridDim.x * (nthr >> 5);
  const int  ntiles = N >> 4;

  for (int t = wave; t < ntiles; t += nwaves) {
    // Opaque zero offset (SGPR): blocks LICM from hoisting the per-layer
    // LDS weight/bias loads out of the loop (prevents the ~512-VGPR +
    // scratch-spill blowup seen when the compiler register-caches weights).
    unsigned loff = 0;
    asm volatile("" : "+s"(loff));
    const _Float16* wS = sW  + loff;
    const float*    bS = sB  + loff;
    const _Float16* bH = sBH + loff;

    const int p = (t << 4) + (lane & 15);
    const float3 P = *(const float3*)(xyz + 3 * p);   // global_load_b96
    const float x = P.x, y = P.y, z = P.z;

    // ---- hash-grid encode straight into layer-1 B fragment ----
    // lo lanes: levels 0..7 ; hi lanes: levels 8..15 ; half 2j/2j+1 = e0/e1
    U16 benc;
#pragma unroll
    for (int j = 0; j < 8; ++j) {
      const int R = lo ? RES_TAB[j] : RES_TAB[j + 8];
      // byte offset of this level's table (4 MB per level), folded into the
      // 32-bit voffset so gathers use uniform-base (SADDR) addressing
      const unsigned lvlbase = lo ? ((unsigned)j << 22) : ((unsigned)(j + 8) << 22);
      const float hR = 0.5f * (float)(R - 1);
      const float sx = fmaf(x, hR, hR);          // (x+1) * 0.5 * (R-1)
      const float sy = fmaf(y, hR, hR);
      const float sz = fmaf(z, hR, hR);
      const float fx = floorf(sx), fy = floorf(sy), fz = floorf(sz);
      const int ix = (int)fx, iy = (int)fy, iz = (int)fz;
      const float tx = sx - fx, ty = sy - fy, tz = sz - fz;
      const int Ri = R - 1;

      // incremental hash: clamp 6 coords once, 4 hash products, 4 xor combos
      const unsigned hx0 = (unsigned)min(max(ix,     0), Ri);
      const unsigned hx1 = (unsigned)min(max(ix + 1, 0), Ri);
      const unsigned hy0 = (unsigned)min(max(iy,     0), Ri) * 2654435761u;
      const unsigned hy1 = (unsigned)min(max(iy + 1, 0), Ri) * 2654435761u;
      const unsigned hz0 = (unsigned)min(max(iz,     0), Ri) * 805459861u;
      const unsigned hz1 = (unsigned)min(max(iz + 1, 0), Ri) * 805459861u;
      const unsigned hyz[4] = { hy0 ^ hz0, hy0 ^ hz1, hy1 ^ hz0, hy1 ^ hz1 };
      const float wy0 = 1.f - ty, wz0 = 1.f - tz;
      const float wyz[4] = { wy0 * wz0, wy0 * tz, ty * wz0, ty * tz };

      // issue all 8 corner gathers first so they can overlap in flight
      float2 emb[8];
#pragma unroll
      for (int c = 0; c < 8; ++c) {
        const unsigned h = ((c & 4) ? hx1 : hx0) ^ hyz[c & 3];
        const unsigned off = lvlbase + ((h & HASH_TM) << 3);
        emb[c] = *(const float2*)((const char*)tables + off);  // L2-resident gather
      }
      float e0 = 0.f, e1 = 0.f;
#pragma unroll
      for (int c = 0; c < 8; ++c) {
        const float w = ((c & 4) ? tx : 1.f - tx) * wyz[c & 3];
        e0 = fmaf(w, emb[c].x, e0);
        e1 = fmaf(w, emb[c].y, e1);
      }
      benc.h[2 * j]     = (_Float16)e0;
      benc.h[2 * j + 1] = (_Float16)e1;
    }

    // ---- density MLP: 32 -> 64 -> 64 -> 16 ----
    D4 dt[4];
    U16 bf[2];
    run_layer_h<1, 4>(wS + 0,    bH + 0,  &benc, dt, lane);  //  4 wmma (f16 acc)
    repack_relu_h<2>(dt, bf, lo);
    run_layer_h<2, 4>(wS + 2048, bH + 64, bf,    dt, lane);  //  8 wmma (f16 acc)
    repack_relu_h<2>(dt, bf, lo);
    C8 cf[1];
    run_layer_f<2, 1>(wS + 6144, bS + 128, bf,   cf, lane);  //  2 wmma (f32, linear)

    // ---- color-input B fragment: 16 feats, K 16..31 zero, no relu ----
    U16 bcol;
#pragma unroll
    for (int j = 0; j < 8; ++j) {
      float v  = cf[0].f[j];
      float vx = xor16f(v);
      bcol.h[j]     = (_Float16)(lo ? v  : 0.f);
      bcol.h[8 + j] = (_Float16)(lo ? vx : 0.f);
    }

    // ---- color MLP: 16 -> 64 -> 64 -> 64 -> 3 ----
    run_layer_h<1, 4>(wS + 7168,  bH + 144, &bcol, dt, lane); //  4 wmma (f16 acc)
    repack_relu_h<2>(dt, bf, lo);
    run_layer_h<2, 4>(wS + 9216,  bH + 208, bf,    dt, lane); //  8 wmma (f16 acc)
    repack_relu_h<2>(dt, bf, lo);
    run_layer_h<2, 4>(wS + 13312, bH + 272, bf,    dt, lane); //  8 wmma (f16 acc)
    repack_relu_h<2>(dt, bf, lo);
    C8 cc[1];
    run_layer_f<2, 1>(wS + 17408, bS + 336, bf,    cc, lane); //  2 wmma (f32)

    // ---- outputs: color (N,3) then sigma (N,1), f32 ----
    if (lo) {
      out[3 * N + p] = expf(cf[0].f[0]);                   // sigma = exp(f[:,0])
      out[3 * p + 0] = 1.f / (1.f + expf(-cc[0].f[0]));
      out[3 * p + 1] = 1.f / (1.f + expf(-cc[0].f[1]));
      out[3 * p + 2] = 1.f / (1.f + expf(-cc[0].f[2]));
    }
  }
}

extern "C" void kernel_launch(void* const* d_in, const int* in_sizes, int n_in,
                              void* d_out, int out_size, void* d_ws, size_t ws_size,
                              hipStream_t stream) {
  const float* xyz    = (const float*)d_in[0];
  const float* tables = (const float*)d_in[1];
  const float* w1  = (const float*)d_in[2];  const float* b1  = (const float*)d_in[3];
  const float* w2  = (const float*)d_in[4];  const float* b2  = (const float*)d_in[5];
  const float* w3  = (const float*)d_in[6];  const float* b3  = (const float*)d_in[7];
  const float* cw1 = (const float*)d_in[8];  const float* cb1 = (const float*)d_in[9];
  const float* cw2 = (const float*)d_in[10]; const float* cb2 = (const float*)d_in[11];
  const float* cw3 = (const float*)d_in[12]; const float* cb3 = (const float*)d_in[13];
  const float* cw4 = (const float*)d_in[14]; const float* cb4 = (const float*)d_in[15];
  float* out = (float*)d_out;
  const int N = in_sizes[0] / 3;            // 1048576 points

  dim3 block(256);                          // 8 wave32 per block
  dim3 grid(2048);                          // 16384 waves -> 4 tiles each
  nerf_fused_kernel<<<grid, block, 0, stream>>>(
      xyz, tables, w1, b1, w2, b2, w3, b3,
      cw1, cb1, cw2, cb2, cw3, cb3, cw4, cb4, out, N);
}
